// HOTFormerBlock_69681549410359
// MI455X (gfx1250) — matrix operand order repaired and checked by reading, not compiled, not run on hardware
//
#include <hip/hip_runtime.h>
#include <hip/hip_bf16.h>
#include <math.h>

// ---- problem constants (from reference) ----
#define M_WIN   8192
#define K_TOK   32
#define C_DIM   256
#define H_HEADS 8
#define DH      32
#define T_TOK   33                  // K + G
#define ROWS    (M_WIN * T_TOK)     // 270336 = 128 * 2112
#define N_PTS   (M_WIN * K_TOK)     // 262144
#define POS_BNDI 25
#define RPE_NUMI 51
#define NEIGH_N 27
#define HID_DIM 1024

typedef __bf16 bf16_t;
typedef __attribute__((ext_vector_type(16))) __bf16 v16bf;
typedef __attribute__((ext_vector_type(8)))  __bf16 v8bf;
typedef __attribute__((ext_vector_type(8)))  float  v8f;
typedef __attribute__((ext_vector_type(4)))  int    v4i;

__device__ __forceinline__ bf16_t f2bf(float f) {
    unsigned u = __builtin_bit_cast(unsigned, f);
    unsigned r = u + 0x7fffu + ((u >> 16) & 1u);   // RNE
    unsigned short h = (unsigned short)(r >> 16);
    return __builtin_bit_cast(bf16_t, h);
}

// ---- CDNA5 async global->LDS copy (ASYNCcnt-tracked), builtin or asm ----
#if __has_builtin(__builtin_amdgcn_global_load_async_to_lds_b128)
#define ASYNC_CP16(dst, src)                                                   \
    __builtin_amdgcn_global_load_async_to_lds_b128(                            \
        (__attribute__((address_space(1))) v4i*)(src),                         \
        (__attribute__((address_space(3))) v4i*)(dst), 0, 0)
#else
#define ASYNC_CP16(dst, src)                                                   \
    asm volatile("global_load_async_to_lds_b128 %0, %1, off" ::                \
                     "v"((unsigned)(unsigned long long)(dst)),                 \
                     "v"((unsigned long long)(src))                            \
                 : "memory")
#endif

#if __has_builtin(__builtin_amdgcn_s_wait_asynccnt)
#define WAIT_ASYNC(n) __builtin_amdgcn_s_wait_asynccnt(n)
#else
#define WAIT_ASYNC(n) asm volatile("s_wait_asynccnt %0" ::"i"(n) : "memory")
#endif

// ------------------------------------------------------------------
// Kernel 1: copy register tokens rt -> xcat row 0 of each window
// ------------------------------------------------------------------
__global__ __launch_bounds__(256) void k_rt_copy(const float* __restrict__ rt,
                                                 float* __restrict__ xcat) {
    size_t i = (size_t)blockIdx.x * 256 + threadIdx.x;   // M_WIN*C threads
    size_t m = i >> 8; int c = (int)(i & 255);
    xcat[(m * T_TOK) * C_DIM + c] = rt[i];
}

// ------------------------------------------------------------------
// Kernel 2: CPE neighbor gather + per-channel einsum, write xcat rows 1..32
// ------------------------------------------------------------------
__global__ __launch_bounds__(256) void k_cpe(const float* __restrict__ data,
                                             const int* __restrict__ neighbors,
                                             const float* __restrict__ cpe_w,
                                             const float* __restrict__ cpe_scale,
                                             const float* __restrict__ cpe_bias,
                                             float* __restrict__ xcat) {
    __shared__ int nbr[NEIGH_N];
    const int n = blockIdx.x;
    const int c = threadIdx.x;
    if (c < NEIGH_N) nbr[c] = neighbors[(size_t)n * NEIGH_N + c];
    __syncthreads();
    float acc = 0.f;
#pragma unroll
    for (int j = 0; j < NEIGH_N; ++j)
        acc += data[(size_t)nbr[j] * C_DIM + c] * cpe_w[j * C_DIM + c];
    float v = data[(size_t)n * C_DIM + c] + acc * cpe_scale[c] + cpe_bias[c];
    int m = n >> 5, kt = n & 31;
    xcat[((size_t)m * T_TOK + 1 + kt) * C_DIM + c] = v;
}

// ------------------------------------------------------------------
// Kernel 3: LayerNorm over C=256 (one block per row), bf16 output
// ------------------------------------------------------------------
__global__ __launch_bounds__(256) void k_ln(const float* __restrict__ x,
                                            const float* __restrict__ g,
                                            const float* __restrict__ b,
                                            bf16_t* __restrict__ y) {
    __shared__ float rs[256], rq[256];
    __shared__ float s_mean, s_inv;
    const int tid = threadIdx.x;
    const size_t row = blockIdx.x;
    float v = x[row * C_DIM + tid];
    rs[tid] = v; rq[tid] = v * v;
    __syncthreads();
    for (int off = 128; off > 0; off >>= 1) {
        if (tid < off) { rs[tid] += rs[tid + off]; rq[tid] += rq[tid + off]; }
        __syncthreads();
    }
    if (tid == 0) {
        float mu = rs[0] * (1.0f / C_DIM);
        float var = rq[0] * (1.0f / C_DIM) - mu * mu;
        s_mean = mu; s_inv = rsqrtf(var + 1e-5f);
    }
    __syncthreads();
    y[row * C_DIM + tid] = f2bf((v - s_mean) * s_inv * g[tid] + b[tid]);
}

// ------------------------------------------------------------------
// Kernel 4: weight convert + transpose: dst[n][k] = bf16(src[k][n])
// ------------------------------------------------------------------
__global__ __launch_bounds__(256) void k_wconv(const float* __restrict__ src,
                                               bf16_t* __restrict__ dst,
                                               int Kd, int Nd) {
    int i = blockIdx.x * 256 + threadIdx.x;
    if (i >= Kd * Nd) return;
    int n = i / Kd, k = i - n * Kd;
    dst[(size_t)n * Kd + k] = f2bf(src[(size_t)k * Nd + n]);
}

// ------------------------------------------------------------------
// Kernel 5: bf16 WMMA GEMM with async double-buffered LDS staging.
//   A:  bf16 row-major [rows x Kd]
//   Bt: bf16 [Nd x Kd]  (pre-transposed weights)
//   block = 128 threads (4 waves); block tile 128x64; K step 32
//   per wave: 32 rows x 64 cols = 8 x v_wmma_f32_16x16x32_bf16 per K step
// ------------------------------------------------------------------
enum { EPI_BIAS = 0, EPI_GELU = 1, EPI_RES = 2 };

template <int EPI, int OUTBF>
__global__ __launch_bounds__(128) void gemm_wmma(const bf16_t* __restrict__ A,
                                                 const bf16_t* __restrict__ Bt,
                                                 const float* __restrict__ bias,
                                                 const float* __restrict__ res,
                                                 void* __restrict__ CoutV,
                                                 int Kd, int Nd) {
    __shared__ __align__(128) bf16_t sA[2][128][32];   // 16 KB
    __shared__ __align__(128) bf16_t sB[2][64][32];    //  8 KB
    const int tid   = threadIdx.x;
    const int lane  = tid & 31;
    const int wave  = tid >> 5;
    const int l15   = lane & 15;
    const int lhalf = lane >> 4;
    const int rBase = blockIdx.y * 128;
    const int cBase = blockIdx.x * 64;

    v8f acc[2][4] = {};

    // issue one K-step tile (A: 4 chunks/thread, B: 2 chunks/thread => 6 async)
    auto issue_tile = [&](int buf, int kk) {
#pragma unroll
        for (int i = 0; i < 4; ++i) {
            int chunk = tid + i * 128;            // 0..511 -> 128x32 bf16
            int row = chunk >> 2;
            int cb  = (chunk & 3) * 16;
            const char* src =
                (const char*)(A + (size_t)(rBase + row) * Kd + kk) + cb;
            char* dst = (char*)&sA[buf][row][0] + cb;
            ASYNC_CP16(dst, src);
        }
#pragma unroll
        for (int i = 0; i < 2; ++i) {
            int chunk = tid + i * 128;            // 0..255 -> 64x32 bf16
            int row = chunk >> 2;
            int cb  = (chunk & 3) * 16;
            const char* src =
                (const char*)(Bt + (size_t)(cBase + row) * Kd + kk) + cb;
            char* dst = (char*)&sB[buf][row][0] + cb;
            ASYNC_CP16(dst, src);
        }
    };

    const int nk = Kd >> 5;
    issue_tile(0, 0);
    for (int s = 0; s < nk; ++s) {
        const int cur = s & 1;
        const bool more = (s + 1) < nk;
        if (more) issue_tile(cur ^ 1, (s + 1) * 32);
        if (more) WAIT_ASYNC(6); else WAIT_ASYNC(0);
        __syncthreads();

        // A fragments (16x32 bf16): lane half -> K {0..7,16..23}/{8..15,24..31}
        const int koff = lhalf * 8;
        union U { v16bf v; v8bf h[2]; };
        U a0, a1;
        a0.h[0] = *(const v8bf*)&sA[cur][wave * 32 + l15][koff];
        a0.h[1] = *(const v8bf*)&sA[cur][wave * 32 + l15][16 + koff];
        a1.h[0] = *(const v8bf*)&sA[cur][wave * 32 + 16 + l15][koff];
        a1.h[1] = *(const v8bf*)&sA[cur][wave * 32 + 16 + l15][16 + koff];
#pragma unroll
        for (int nt = 0; nt < 4; ++nt) {
            // B fragment (32x16 bf16): lane half -> K {0..7,16..23}/{8..15,24..31}
            U bu;
            bu.h[0] = *(const v8bf*)&sB[cur][nt * 16 + l15][koff];
            bu.h[1] = *(const v8bf*)&sB[cur][nt * 16 + l15][16 + koff];
            v16bf bfr = bu.v;
            acc[0][nt] = __builtin_amdgcn_wmma_f32_16x16x32_bf16(
                false, a0.v, false, bfr, (short)0, acc[0][nt], false, false);
            acc[1][nt] = __builtin_amdgcn_wmma_f32_16x16x32_bf16(
                false, a1.v, false, bfr, (short)0, acc[1][nt], false, false);
        }
        __syncthreads();
    }

    // epilogue: D layout = VGPR r -> row r (lanes 0-15) / r+8 (lanes 16-31)
#pragma unroll
    for (int mi = 0; mi < 2; ++mi) {
        const int rowOff = rBase + wave * 32 + mi * 16 + lhalf * 8;
#pragma unroll
        for (int nt = 0; nt < 4; ++nt) {
            const int col = cBase + nt * 16 + l15;
            const float bv = bias[col];
#pragma unroll
            for (int r = 0; r < 8; ++r) {
                const size_t row = (size_t)(rowOff + r);
                float v = acc[mi][nt][r] + bv;
                if (EPI == EPI_GELU)
                    v = 0.5f * v * (1.0f + erff(v * 0.70710678118654752f));
                if (EPI == EPI_RES)
                    v += res[row * Nd + col];
                if (OUTBF)
                    ((bf16_t*)CoutV)[row * Nd + col] = f2bf(v);
                else
                    ((float*)CoutV)[row * Nd + col] = v;
            }
        }
    }
}

// ------------------------------------------------------------------
// Kernel 6: attention per (m, h): scores + RPE + mask + softmax + AV
// ------------------------------------------------------------------
__global__ __launch_bounds__(128) void k_attn(const float* __restrict__ qkv,
                                              const int* __restrict__ rel_pos,
                                              const float* __restrict__ mask,
                                              const float* __restrict__ rpe_table,
                                              bf16_t* __restrict__ out) {
    const int m = blockIdx.x >> 3;
    const int h = blockIdx.x & 7;
    __shared__ float sq[T_TOK][DH], sk[T_TOK][DH], sv[T_TOK][DH];
    __shared__ float ss[T_TOK][T_TOK + 1];
    const int tid = threadIdx.x;
    const float scale = 0.17677669529663687f;   // (C/H)^-0.5 = 1/sqrt(32)

    for (int i = tid; i < T_TOK * DH; i += 128) {
        int t = i >> 5, d = i & 31;
        size_t base = ((size_t)m * T_TOK + t) * (3 * C_DIM) + h * DH + d;
        sq[t][d] = qkv[base] * scale;
        sk[t][d] = qkv[base + C_DIM];
        sv[t][d] = qkv[base + 2 * C_DIM];
    }
    __syncthreads();

    for (int i = tid; i < T_TOK * T_TOK; i += 128) {
        int qi = i / T_TOK, ki = i % T_TOK;
        float s = 0.f;
#pragma unroll
        for (int d = 0; d < DH; ++d) s += sq[qi][d] * sk[ki][d];
        if (qi > 0 && ki > 0) {
            const int* rp =
                &rel_pos[(((size_t)m * K_TOK + (qi - 1)) * K_TOK + (ki - 1)) * 3];
#pragma unroll
            for (int t = 0; t < 3; ++t) {
                int r = rp[t];
                r = r < -POS_BNDI ? -POS_BNDI : (r > POS_BNDI ? POS_BNDI : r);
                s += rpe_table[(size_t)(r + POS_BNDI + t * RPE_NUMI) * H_HEADS + h];
            }
        }
        s += mask[((size_t)m * T_TOK + qi) * T_TOK + ki];
        ss[qi][ki] = s;
    }
    __syncthreads();

    if (tid < T_TOK) {
        float mx = -1e30f;
        for (int ki = 0; ki < T_TOK; ++ki) mx = fmaxf(mx, ss[tid][ki]);
        float sum = 0.f;
        for (int ki = 0; ki < T_TOK; ++ki) {
            float e = expf(ss[tid][ki] - mx);
            ss[tid][ki] = e; sum += e;
        }
        float inv = 1.f / sum;
        for (int ki = 0; ki < T_TOK; ++ki) ss[tid][ki] *= inv;
    }
    __syncthreads();

    for (int i = tid; i < T_TOK * DH; i += 128) {
        int qi = i >> 5, d = i & 31;
        float s = 0.f;
        for (int ki = 0; ki < T_TOK; ++ki) s += ss[qi][ki] * sv[ki][d];
        out[((size_t)m * T_TOK + qi) * C_DIM + h * DH + d] = f2bf(s);
    }
}

// ------------------------------------------------------------------
// Kernel 7: split xcat -> (data_out [N,C], rt_out [M,C]) concatenated
// ------------------------------------------------------------------
__global__ __launch_bounds__(256) void k_split(const float* __restrict__ xcat,
                                               float* __restrict__ out) {
    size_t i = (size_t)blockIdx.x * 256 + threadIdx.x;
    const size_t ND = (size_t)N_PTS * C_DIM;
    if (i < ND) {
        size_t n = i >> 8; int c = (int)(i & 255);
        size_t m = n >> 5; int kt = (int)(n & 31);
        out[i] = xcat[(m * T_TOK + 1 + kt) * C_DIM + c];
    } else {
        size_t j = i - ND;
        size_t m = j >> 8; int c = (int)(j & 255);
        out[i] = xcat[(m * T_TOK) * C_DIM + c];
    }
}

// ------------------------------------------------------------------
extern "C" void kernel_launch(void* const* d_in, const int* in_sizes, int n_in,
                              void* d_out, int out_size, void* d_ws, size_t ws_size,
                              hipStream_t stream) {
    const float* data      = (const float*)d_in[0];
    const float* rt        = (const float*)d_in[1];
    const int*   neighbors = (const int*)  d_in[2];
    const int*   rel_pos   = (const int*)  d_in[3];
    const float* mask      = (const float*)d_in[4];
    const float* cpe_w     = (const float*)d_in[5];
    const float* cpe_scale = (const float*)d_in[6];
    const float* cpe_bias  = (const float*)d_in[7];
    const float* ln1_g     = (const float*)d_in[8];
    const float* ln1_b     = (const float*)d_in[9];
    const float* qkv_w     = (const float*)d_in[10];
    const float* qkv_b     = (const float*)d_in[11];
    const float* rpe_table = (const float*)d_in[12];
    const float* proj_w    = (const float*)d_in[13];
    const float* proj_b    = (const float*)d_in[14];
    const float* ln2_g     = (const float*)d_in[15];
    const float* ln2_b     = (const float*)d_in[16];
    const float* mlp_w1    = (const float*)d_in[17];
    const float* mlp_b1    = (const float*)d_in[18];
    const float* mlp_w2    = (const float*)d_in[19];
    const float* mlp_b2    = (const float*)d_in[20];

    const size_t szXcat = (size_t)ROWS * C_DIM * 4;        // fp32 residual chain
    const size_t szHbuf = (size_t)ROWS * C_DIM * 2;        // bf16 LN out
    const size_t szQkv  = (size_t)ROWS * (3 * C_DIM) * 4;  // fp32 qkv (reused: bf16 hid)
    const size_t szAttn = (size_t)ROWS * C_DIM * 2;        // bf16 attention out
    char* w = (char*)d_ws;
    float*  xcat  = (float*)(w);
    bf16_t* hbuf  = (bf16_t*)(w + szXcat);
    float*  qkv   = (float*)(w + szXcat + szHbuf);
    bf16_t* hid   = (bf16_t*)(w + szXcat + szHbuf);                 // reuse qkv region
    bf16_t* attnb = (bf16_t*)(w + szXcat + szHbuf + szQkv);
    char*   wts   = w + szXcat + szHbuf + szQkv + szAttn;
    bf16_t* qkv_wt  = (bf16_t*)(wts);                               // [768][256]
    bf16_t* proj_wt = (bf16_t*)(wts + (size_t)768 * 256 * 2);       // [256][256]
    bf16_t* w1t     = (bf16_t*)(wts + (size_t)(768 + 256) * 256 * 2);        // [1024][256]
    bf16_t* w2t     = (bf16_t*)(wts + (size_t)(768 + 256 + 1024) * 256 * 2); // [256][1024]

    // weight convert+transpose (tiny, once per launch)
    k_wconv<<<(C_DIM * 3 * C_DIM + 255) / 256, 256, 0, stream>>>(qkv_w, qkv_wt, C_DIM, 3 * C_DIM);
    k_wconv<<<(C_DIM * C_DIM + 255) / 256, 256, 0, stream>>>(proj_w, proj_wt, C_DIM, C_DIM);
    k_wconv<<<(C_DIM * HID_DIM + 255) / 256, 256, 0, stream>>>(mlp_w1, w1t, C_DIM, HID_DIM);
    k_wconv<<<(HID_DIM * C_DIM + 255) / 256, 256, 0, stream>>>(mlp_w2, w2t, HID_DIM, C_DIM);

    // 1) register tokens into row 0 of each window
    k_rt_copy<<<(M_WIN * C_DIM) / 256, 256, 0, stream>>>(rt, xcat);
    // 2) CPE gather + residual into rows 1..32
    k_cpe<<<N_PTS, 256, 0, stream>>>(data, neighbors, cpe_w, cpe_scale, cpe_bias, xcat);
    // 3) LN1 -> bf16
    k_ln<<<ROWS, 256, 0, stream>>>(xcat, ln1_g, ln1_b, hbuf);
    // 4) QKV GEMM (270336 x 256 x 768) -> fp32 qkv
    gemm_wmma<EPI_BIAS, 0><<<dim3((3 * C_DIM) / 64, ROWS / 128), 128, 0, stream>>>(
        hbuf, qkv_wt, qkv_b, nullptr, qkv, C_DIM, 3 * C_DIM);
    // 5) attention per (m, h) -> bf16
    k_attn<<<M_WIN * H_HEADS, 128, 0, stream>>>(qkv, rel_pos, mask, rpe_table, attnb);
    // 6) proj GEMM + residual (in-place over xcat, fp32)
    gemm_wmma<EPI_RES, 0><<<dim3(C_DIM / 64, ROWS / 128), 128, 0, stream>>>(
        attnb, proj_wt, proj_b, xcat, xcat, C_DIM, C_DIM);
    // 7) LN2 -> bf16
    k_ln<<<ROWS, 256, 0, stream>>>(xcat, ln2_g, ln2_b, hbuf);
    // 8) MLP up-projection + exact GELU -> bf16 hid (reuses qkv region)
    gemm_wmma<EPI_GELU, 1><<<dim3(HID_DIM / 64, ROWS / 128), 128, 0, stream>>>(
        hbuf, w1t, mlp_b1, nullptr, hid, C_DIM, HID_DIM);
    // 9) MLP down-projection + residual (in-place over xcat, fp32)
    gemm_wmma<EPI_RES, 0><<<dim3(C_DIM / 64, ROWS / 128), 128, 0, stream>>>(
        hid, w2t, mlp_b2, xcat, xcat, HID_DIM, C_DIM);
    // 10) split outputs: data_out then rt_out
    k_split<<<(N_PTS * C_DIM + M_WIN * C_DIM) / 256, 256, 0, stream>>>(xcat, (float*)d_out);
}
